// GATEDGNN_48318382080246
// MI455X (gfx1250) — matrix operand order repaired and checked by reading, not compiled, not run on hardware
//
#include <hip/hip_runtime.h>
#include <hip/hip_bf16.h>
#include <math.h>

#define NNODE 50000L
#define NEDGE 625000L
#define DIM   128
#define NLAYER 4
#define NBOND_TAB (37L * DIM)

typedef __attribute__((ext_vector_type(16))) _Float16 v16h;
typedef __attribute__((ext_vector_type(8)))  float    v8f;

__device__ __forceinline__ float gelu_tanh(float x) {
  const float c = 0.7978845608028654f; // sqrt(2/pi)
  float t = tanhf(c * (x + 0.044715f * x * x * x));
  return 0.5f * x * (1.0f + t);
}

// ---------------------------------------------------------------------------
// Pack weights (L x 5 mats x 128 x 128 f32, row-major KxN) into WMMA B-fragment
// layout, f16. Per weight: 32 tiles (kt 0..3, nt 0..7), each tile = 32 lanes x
// 16 halves contiguous. Lane holds column n = nt*16 + (lane&15);
// half i -> K = kt*32 + 16*(lane>>4) + i.
// mat order within a layer: 0=WA 1=WB 2=WD 3=WE 4=WC
// ---------------------------------------------------------------------------
__global__ void pack_w_kernel(const float* __restrict__ WA, const float* __restrict__ WB,
                              const float* __restrict__ WD, const float* __restrict__ WE,
                              const float* __restrict__ WC, _Float16* __restrict__ pw,
                              int total) {
  int idx = blockIdx.x * blockDim.x + threadIdx.x;
  if (idx >= total) return;
  int i    = idx & 15;
  int lane = (idx >> 4) & 31;
  int tile = (idx >> 9) & 31;
  int lm   = idx >> 14;            // l*5 + mat
  int l = lm / 5, mat = lm % 5;
  int kt = tile >> 3, nt = tile & 7;
  int k = kt * 32 + ((lane >> 4) << 4) + i;
  int n = (nt << 4) + (lane & 15);
  const float* W = (mat == 0) ? WA : (mat == 1) ? WB : (mat == 2) ? WD
                 : (mat == 3) ? WE : WC;
  pw[idx] = (_Float16)W[(long)l * DIM * DIM + (long)k * DIM + n];
}

// ---------------------------------------------------------------------------
// BondEncoder (float4-vectorized): e[e, 4d..4d+3] = sum_f bond_emb[f][attr][...]
// ---------------------------------------------------------------------------
__global__ void bond_embed_kernel(const int* __restrict__ edge_attr,
                                  const float4* __restrict__ bond_emb4,
                                  float4* __restrict__ ebuf4, long total4) {
  long idx = (long)blockIdx.x * blockDim.x + threadIdx.x;
  if (idx >= total4) return;
  long e = idx >> 5;               // 32 float4 groups per edge row
  int  g = (int)(idx & 31);
  int a0 = edge_attr[e * 3 + 0];
  int a1 = edge_attr[e * 3 + 1];
  int a2 = edge_attr[e * 3 + 2];
  const int R = DIM / 4;           // 32 float4 per table row
  float4 v0 = bond_emb4[(long)a0 * R + g];
  float4 v1 = bond_emb4[37L * R + (long)a1 * R + g];
  float4 v2 = bond_emb4[2 * 37L * R + (long)a2 * R + g];
  float4 o;
  o.x = v0.x + v1.x + v2.x;
  o.y = v0.y + v1.y + v2.y;
  o.z = v0.z + v1.z + v2.z;
  o.w = v0.w + v1.w + v2.w;
  ebuf4[idx] = o;
}

// ---------------------------------------------------------------------------
// WMMA GEMM: O[m] = A(Mx128, f32) @ W[m](128x128, packed f16) + bias[m]
// One wave computes a 16-row strip across all 8 column tiles for NMAT weights.
// NMAT is a template arg so all output/bias pointers are compile-time selected
// (global_store, no s_cselect / flat paths). Stores take a wave-uniform
// full-tile fast path (no EXEC churn) except for the single ragged edge tile.
// ---------------------------------------------------------------------------
template <int NMAT>
__global__ void gemm_wmma_kernel(const float* __restrict__ A, long M, long Mtiles,
                                 const _Float16* __restrict__ Wp,
                                 float* __restrict__ O0, float* __restrict__ O1,
                                 float* __restrict__ O2, float* __restrict__ O3,
                                 const float* __restrict__ b0, const float* __restrict__ b1,
                                 const float* __restrict__ b2, const float* __restrict__ b3) {
  long wave = ((long)blockIdx.x * blockDim.x + threadIdx.x) >> 5;
  if (wave >= Mtiles) return;
  int lane = threadIdx.x & 31;
  int half = lane >> 4;
  long row0 = wave << 4;
  long m = row0 + (lane & 15);
  if (m >= M) m = M - 1;               // clamp load row (stores handle ragged)
  const float* ap = A + m * DIM;
  const bool full_tile = (row0 + 16 <= M);   // wave-uniform

  v16h a[4];
#pragma unroll
  for (int kt = 0; kt < 4; kt++) {
    const float* p = ap + kt * 32 + half * 16;
#pragma unroll
    for (int i = 0; i < 8; i++) a[kt][i] = (_Float16)p[i];
#pragma unroll
    for (int i = 0; i < 8; i++) a[kt][8 + i] = (_Float16)p[8 + i];
  }

  float* Os[4] = {O0, O1, O2, O3};
  const float* Bs[4] = {b0, b1, b2, b3};
#pragma unroll
  for (int mm = 0; mm < NMAT; mm++) {
    const _Float16* w = Wp + (long)mm * (DIM * DIM);
    float* O = Os[mm];
    const float* bias = Bs[mm];
#pragma unroll
    for (int nt = 0; nt < 8; nt++) {
      v8f acc = {};
#pragma unroll
      for (int kt = 0; kt < 4; kt++) {
        v16h b = *(const v16h*)(w + ((kt * 8 + nt) * 512) + lane * 16);
        acc = __builtin_amdgcn_wmma_f32_16x16x32_f16(
            /*neg_a=*/false, a[kt], /*neg_b=*/false, b,
            /*c_mod=*/(short)0, acc, /*reuse_a=*/false, /*reuse_b=*/false);
      }
      int n = (nt << 4) + (lane & 15);
      float bv = bias[n];
      float* op = O + (row0 + half * 8) * DIM + n;
      if (full_tile) {
#pragma unroll
        for (int r = 0; r < 8; r++) op[r * DIM] = acc[r] + bv;
      } else {
#pragma unroll
        for (int r = 0; r < 8; r++) {
          if (row0 + half * 8 + r < M) op[r * DIM] = acc[r] + bv;
        }
      }
    }
  }
}

// ---------------------------------------------------------------------------
// Edge message: e_ij = Ce(in eij) + Dx[dst] + Ex[src] (in place); sigma gate;
// scatter num/den via f32 atomics (L2-resident); BN partial sums for e_ij.
// Block = 128 channels x EPB edges; src/dst staged once through LDS.
// ---------------------------------------------------------------------------
#define EPB 64
__global__ void edge_msg_kernel(const int* __restrict__ eidx,
                                const float* __restrict__ Bx, const float* __restrict__ Dx,
                                const float* __restrict__ Ex, float* __restrict__ eij,
                                float* __restrict__ num, float* __restrict__ den,
                                float* __restrict__ esum, float* __restrict__ esq) {
  __shared__ int ss[EPB];
  __shared__ int tt[EPB];
  int d = threadIdx.x;
  long e0 = (long)blockIdx.x * EPB;
  {
    int j = threadIdx.x & (EPB - 1);
    long e = e0 + j;
    if (e >= NEDGE) e = NEDGE - 1;       // clamped (loop below never uses it)
    if (threadIdx.x < EPB) ss[j] = eidx[e];
    else                   tt[j] = eidx[NEDGE + e];
  }
  __syncthreads();
  float ls = 0.f, lq = 0.f;
  int jmax = (int)((NEDGE - e0) < EPB ? (NEDGE - e0) : EPB);
  for (int j = 0; j < jmax; j++) {
    long e = e0 + j;
    int s = ss[j];            // source
    int t = tt[j];            // target / scatter index
    long ei = e * DIM + d;
    float v = eij[ei] + Dx[(long)t * DIM + d] + Ex[(long)s * DIM + d];
    eij[ei] = v;
    float sg = 1.f / (1.f + expf(-v));
    atomicAdd(&num[(long)t * DIM + d], sg * Bx[(long)s * DIM + d]);
    atomicAdd(&den[(long)t * DIM + d], sg);
    ls += v;
    lq += v * v;
  }
  atomicAdd(&esum[d], ls);
  atomicAdd(&esq[d], lq);
}

// ---------------------------------------------------------------------------
// Node aggregate: x_new = Ax + num/(den+1e-6) (in place over Ax) + BN stats.
// ---------------------------------------------------------------------------
#define NPB 64
__global__ void node_agg_kernel(float* __restrict__ Ax, const float* __restrict__ num,
                                const float* __restrict__ den,
                                float* __restrict__ xsum, float* __restrict__ xsq) {
  int d = threadIdx.x;
  long n0 = (long)blockIdx.x * NPB;
  float ls = 0.f, lq = 0.f;
  for (int j = 0; j < NPB; j++) {
    long n = n0 + j;
    if (n >= NNODE) break;
    long idx = n * DIM + d;
    float v = Ax[idx] + num[idx] / (den[idx] + 1e-6f);
    Ax[idx] = v;
    ls += v;
    lq += v * v;
  }
  atomicAdd(&xsum[d], ls);
  atomicAdd(&xsq[d], lq);
}

// ---------------------------------------------------------------------------
// dst += gelu(bn(h)), float4-vectorized (b128 loads/stores on the big streams).
// Training-mode BN: biased variance, eps=1e-5.
// ---------------------------------------------------------------------------
__global__ void apply_bn_gelu_res_kernel(float4* __restrict__ dst, const float4* __restrict__ h,
                                         const float* __restrict__ sum, const float* __restrict__ sq,
                                         const float* __restrict__ g, const float* __restrict__ b,
                                         float inv_n, long total4) {
  long idx = (long)blockIdx.x * blockDim.x + threadIdx.x;
  if (idx >= total4) return;
  int d0 = (int)(idx & 31) * 4;     // 32 float4 groups per 128-channel row
  float4 hv = h[idx];
  float4 dv = dst[idx];
  float o[4] = {hv.x, hv.y, hv.z, hv.w};
  float r[4] = {dv.x, dv.y, dv.z, dv.w};
#pragma unroll
  for (int c = 0; c < 4; c++) {
    int d = d0 + c;
    float mean = sum[d] * inv_n;
    float var  = sq[d] * inv_n - mean * mean;
    float y = g[d] * (o[c] - mean) * rsqrtf(var + 1e-5f) + b[d];
    r[c] += gelu_tanh(y);
  }
  float4 out;
  out.x = r[0]; out.y = r[1]; out.z = r[2]; out.w = r[3];
  dst[idx] = out;
}

// ---------------------------------------------------------------------------
extern "C" void kernel_launch(void* const* d_in, const int* in_sizes, int n_in,
                              void* d_out, int out_size, void* d_ws, size_t ws_size,
                              hipStream_t stream) {
  const float* X_n        = (const float*)d_in[0];
  // d_in[1] = PE (unused: _pe_lap is dead code in the reference)
  const int*   edge_index = (const int*)d_in[2];
  const int*   edge_attr  = (const int*)d_in[3];
  const float* bond_emb   = (const float*)d_in[4];
  const float* WA = (const float*)d_in[5];  const float* bA = (const float*)d_in[6];
  const float* WB = (const float*)d_in[7];  const float* bB = (const float*)d_in[8];
  const float* WC = (const float*)d_in[9];  const float* bC = (const float*)d_in[10];
  const float* WD = (const float*)d_in[11]; const float* bD = (const float*)d_in[12];
  const float* WE = (const float*)d_in[13]; const float* bE = (const float*)d_in[14];
  const float* bn_x_g = (const float*)d_in[15]; const float* bn_x_b = (const float*)d_in[16];
  const float* bn_e_g = (const float*)d_in[17]; const float* bn_e_b = (const float*)d_in[18];
  // d_in[19] W_pe, d_in[20] b_pe: unused

  float* x = (float*)d_out;   // running node features live in d_out

  // ---- workspace carve ----
  char* base = (char*)d_ws;
  size_t off = 0;
  auto carve = [&](size_t bytes) -> char* {
    char* p = base + off;
    off += (bytes + 255) & ~(size_t)255;
    return p;
  };
  float* eij  = (float*)carve(NEDGE * DIM * 4);   // Ce then e_ij
  float* ebuf = (float*)carve(NEDGE * DIM * 4);   // running edge features e
  float* Ax   = (float*)carve(NNODE * DIM * 4);   // also holds x_new
  float* Bx   = (float*)carve(NNODE * DIM * 4);
  float* Dx   = (float*)carve(NNODE * DIM * 4);
  float* Ex   = (float*)carve(NNODE * DIM * 4);
  float* num  = (float*)carve(NNODE * DIM * 4);
  float* den  = (float*)carve(NNODE * DIM * 4);
  _Float16* pw = (_Float16*)carve((size_t)NLAYER * 5 * DIM * DIM * 2);
  float* xsum = (float*)carve(DIM * 4);
  float* xsq  = (float*)carve(DIM * 4);
  float* esum = (float*)carve(DIM * 4);
  float* esq  = (float*)carve(DIM * 4);
  (void)ws_size; (void)n_in; (void)in_sizes; (void)out_size;

  // ---- init: x = X_n; pack weights; bond embedding ----
  hipMemcpyAsync(x, X_n, (size_t)NNODE * DIM * 4, hipMemcpyDeviceToDevice, stream);

  {
    int total = NLAYER * 5 * DIM * DIM;  // 327680
    pack_w_kernel<<<(total + 255) / 256, 256, 0, stream>>>(WA, WB, WD, WE, WC, pw, total);
  }
  {
    long total4 = NEDGE * (DIM / 4);
    bond_embed_kernel<<<(int)((total4 + 255) / 256), 256, 0, stream>>>(
        edge_attr, (const float4*)bond_emb, (float4*)ebuf, total4);
  }

  const long node_tiles = (NNODE + 15) / 16;   // 3125 (exact: 50000 = 3125*16)
  const long edge_tiles = (NEDGE + 15) / 16;   // 39063 (last tile ragged)
  const int  gemm_block = 128;                 // 4 waves/block

  for (int l = 0; l < NLAYER; l++) {
    hipMemsetAsync(num, 0, (size_t)NNODE * DIM * 4, stream);
    hipMemsetAsync(den, 0, (size_t)NNODE * DIM * 4, stream);
    hipMemsetAsync(xsum, 0, DIM * 4, stream);
    hipMemsetAsync(xsq,  0, DIM * 4, stream);
    hipMemsetAsync(esum, 0, DIM * 4, stream);
    hipMemsetAsync(esq,  0, DIM * 4, stream);

    const _Float16* pwl = pw + (size_t)l * 5 * DIM * DIM;

    // node GEMMs: Ax, Bx, Dx, Ex
    gemm_wmma_kernel<4><<<(int)((node_tiles + 3) / 4), gemm_block, 0, stream>>>(
        x, NNODE, node_tiles, pwl, Ax, Bx, Dx, Ex,
        bA + l * DIM, bB + l * DIM, bD + l * DIM, bE + l * DIM);

    // edge GEMM: Ce -> eij
    gemm_wmma_kernel<1><<<(int)((edge_tiles + 3) / 4), gemm_block, 0, stream>>>(
        ebuf, NEDGE, edge_tiles, pwl + 4 * DIM * DIM, eij, eij, eij, eij,
        bC + l * DIM, bC + l * DIM, bC + l * DIM, bC + l * DIM);

    // gather + gate + scatter + e BN stats
    edge_msg_kernel<<<(int)((NEDGE + EPB - 1) / EPB), DIM, 0, stream>>>(
        edge_index, Bx, Dx, Ex, eij, num, den, esum, esq);

    // node aggregate + x BN stats (x_new written into Ax)
    node_agg_kernel<<<(int)((NNODE + NPB - 1) / NPB), DIM, 0, stream>>>(Ax, num, den, xsum, xsq);

    // x += gelu(bn(x_new))
    {
      long total4 = NNODE * (DIM / 4);
      apply_bn_gelu_res_kernel<<<(int)((total4 + 255) / 256), 256, 0, stream>>>(
          (float4*)x, (const float4*)Ax, xsum, xsq,
          bn_x_g + l * DIM, bn_x_b + l * DIM, 1.0f / (float)NNODE, total4);
    }
    // e += gelu(bn(e_ij))
    {
      long total4 = NEDGE * (DIM / 4);
      apply_bn_gelu_res_kernel<<<(int)((total4 + 255) / 256), 256, 0, stream>>>(
          (float4*)ebuf, (const float4*)eij, esum, esq,
          bn_e_g + l * DIM, bn_e_b + l * DIM, 1.0f / (float)NEDGE, total4);
    }
  }
}
